// MultiheadAttention_6064493822545
// MI455X (gfx1250) — matrix operand order repaired and checked by reading, not compiled, not run on hardware
//
#include <hip/hip_runtime.h>
#include <math.h>

typedef __attribute__((ext_vector_type(16))) _Float16 h16;
typedef __attribute__((ext_vector_type(8)))  _Float16 h8;
typedef __attribute__((ext_vector_type(4)))  _Float16 h4;
typedef __attribute__((ext_vector_type(8)))  float    f8;
typedef __attribute__((ext_vector_type(4)))  unsigned int u32x4;
typedef __attribute__((ext_vector_type(8)))  int i32x8;
typedef __attribute__((ext_vector_type(4)))  int i32x4;

#define WMMA_F32_F16(A, B, C) \
    __builtin_amdgcn_wmma_f32_16x16x32_f16(false, (A), false, (B), (short)0, (C), false, false)

#define CAT16(lo, hi) __builtin_shufflevector((lo), (hi), 0,1,2,3,4,5,6,7,8,9,10,11,12,13,14,15)

// Problem dimensions
#define BB 2
#define SS 2048
#define DD 1024
#define HH 16
#define HD 64
#define MT (BB*SS)          // 4096 rows

// ---------------------------------------------------------------------------
// Elementwise f32 -> f16 conversion (vectorized, grid-stride).
// ---------------------------------------------------------------------------
__global__ __launch_bounds__(256) void cvt_f32_f16(
    const float* __restrict__ src, _Float16* __restrict__ dst, int n4)
{
    int i = blockIdx.x * blockDim.x + threadIdx.x;
    const int stride = gridDim.x * blockDim.x;
    for (; i < n4; i += stride) {
        float4 v = reinterpret_cast<const float4*>(src)[i];
        h4 o;
        o[0] = (_Float16)v.x; o[1] = (_Float16)v.y;
        o[2] = (_Float16)v.z; o[3] = (_Float16)v.w;
        reinterpret_cast<h4*>(dst)[i] = o;
    }
}

// ---------------------------------------------------------------------------
// TDM: DMA a 64-row x 32-half tile of W (f16, row stride 1024 halves) into LDS.
// LDS rows are padded to 40 halves (80B): pad_interval = 16 dwords (64B, one
// row), pad_amount = 4 dwords (16B).  D# per CDNA5 ISA ch.8 (2-D tensor form).
// ---------------------------------------------------------------------------
__device__ __forceinline__ void tdm_load_wtile(const _Float16* gsrc, unsigned lds_off)
{
    unsigned long long ga = (unsigned long long)(uintptr_t)gsrc;
    u32x4 g0;
    g0[0] = 1u;                                   // count=1 (valid user D#)
    g0[1] = lds_off;                              // lds_addr (bytes)
    g0[2] = (unsigned)(ga & 0xFFFFFFFFu);         // global_addr[31:0]
    g0[3] = (unsigned)((ga >> 32) & 0x1FFFFFFu)   // global_addr[56:32]
          | (2u << 30);                           // type=2 ("image")
    i32x8 g1;
    g1[0] = (int)((1u << 16)                      // data_size = 2 bytes
                | (1u << 20)                      // pad_enable
                | (3u << 22)                      // pad_interval: 16 dwords
                | (3u << 25));                    // pad_amount: 4 dwords
    g1[1] = (int)(1024u << 16);                   // tensor_dim0 = 1024 (lo16)
    g1[2] = (int)(1024u << 16);                   // tensor_dim1 = 1024 (lo16)
    g1[3] = (int)(32u << 16);                     // tile_dim0 = 32
    g1[4] = 64;                                   // tile_dim1 = 64, tile_dim2 = 0
    g1[5] = 1024;                                 // tensor_dim0_stride = 1024
    g1[6] = 0;
    g1[7] = 0;
    i32x4 z4 = {0, 0, 0, 0};
    i32x8 z8 = {0, 0, 0, 0, 0, 0, 0, 0};
    __builtin_amdgcn_tensor_load_to_lds(g0, g1, z4, z4, z8, 0);
}

// ---------------------------------------------------------------------------
// f16 GEMM: Y = A @ W^T + bias.  A: [MT, DD] f16, W: [DD, DD] f16.
// Block = 4 waves sharing one 64-col N tile; W tiles staged in LDS by the
// Tensor Data Mover (double buffered), A rows streamed per wave from global.
// Each wave: 32x64 tile, 8 WMMA per 32-deep K-step.
//   mode 0: Y f16 at [((b*H+h)*S + s)*HD + d]          (Q, K layout)
//   mode 1: Y f16 at [((b*H+h)*HD + d)*S + s]          (V^T layout)
//   mode 2: Y f32 dense [m*DD + n] + bias              (final output)
// ---------------------------------------------------------------------------
__global__ __launch_bounds__(128) void gemm16_kernel(
    const _Float16* __restrict__ A, const _Float16* __restrict__ W,
    const float* __restrict__ bias, _Float16* __restrict__ Yh,
    float* __restrict__ Yf, int mode)
{
    __shared__ _Float16 wtile[2][64 * 40];   // double-buffered 64x32 tile, 80B rows

    const int lane = threadIdx.x & 31;
    const int wv   = threadIdx.x >> 5;
    const int task = blockIdx.x * 4 + wv;
    const int mt  = task & 127;         // 128 M tiles of 32
    const int nt4 = task >> 7;          // 16 N tiles of 64 (same for all 4 waves)
    const int m0 = mt * 32, n0 = nt4 * 64;
    const int hi = lane >> 4, lm = lane & 15;

    f8 acc[2][4];
#pragma unroll
    for (int mi = 0; mi < 2; ++mi)
#pragma unroll
        for (int n = 0; n < 4; ++n)
#pragma unroll
            for (int v = 0; v < 8; ++v) acc[mi][n][v] = 0.0f;

    const _Float16* Ar0 = A + (size_t)(m0 + lm) * DD;
    const _Float16* Ar1 = A + (size_t)(m0 + 16 + lm) * DD;
    const _Float16* Wt  = W + (size_t)n0 * DD;   // tile origin row

    // Prologue: wave 0 queues the first two W tiles on the TDM
    if (wv == 0) {
        tdm_load_wtile(Wt + 0,  (unsigned)(uintptr_t)&wtile[0][0]);
        tdm_load_wtile(Wt + 32, (unsigned)(uintptr_t)&wtile[1][0]);
    }

    int ib = 0;
    for (int kk = 0; kk < DD; kk += 32, ib ^= 1) {
        // Per-wave A loads overlap the TDM wait
        h8 a0l = *reinterpret_cast<const h8*>(Ar0 + kk + hi * 8);
        h8 a0h = *reinterpret_cast<const h8*>(Ar0 + kk + 16 + hi * 8);
        h8 a1l = *reinterpret_cast<const h8*>(Ar1 + kk + hi * 8);
        h8 a1h = *reinterpret_cast<const h8*>(Ar1 + kk + 16 + hi * 8);
        __builtin_prefetch(Ar0 + kk + 32, 0, 3);
        __builtin_prefetch(Ar1 + kk + 32, 0, 3);

        if (wv == 0) {
            if (kk + 32 < DD) __builtin_amdgcn_s_wait_tensorcnt(1);
            else              __builtin_amdgcn_s_wait_tensorcnt(0);
        }
        __syncthreads();                 // W tile [ib] ready for all waves

        const _Float16* wbuf = &wtile[ib][0];
        h16 bf[4];
#pragma unroll
        for (int nt = 0; nt < 4; ++nt) {
            const int row = nt * 16 + lm;
            h8 b0 = *reinterpret_cast<const h8*>(wbuf + row * 40 + hi * 8);
            h8 b1 = *reinterpret_cast<const h8*>(wbuf + row * 40 + 16 + hi * 8);
            bf[nt] = CAT16(b0, b1);
        }

        h16 af0 = CAT16(a0l, a0h);
        h16 af1 = CAT16(a1l, a1h);
#pragma unroll
        for (int nt = 0; nt < 4; ++nt) {
            acc[0][nt] = WMMA_F32_F16(af0, bf[nt], acc[0][nt]);
            acc[1][nt] = WMMA_F32_F16(af1, bf[nt], acc[1][nt]);
        }

        __syncthreads();                 // all waves done reading buffer [ib]
        if (wv == 0 && kk + 64 < DD)
            tdm_load_wtile(Wt + kk + 64, (unsigned)(uintptr_t)&wtile[ib][0]);
    }

#pragma unroll
    for (int mi = 0; mi < 2; ++mi) {
#pragma unroll
        for (int nt = 0; nt < 4; ++nt) {
            const int ncol = n0 + nt * 16 + lm;
            const float bb = bias[ncol];
            const int hh = ncol >> 6, d = ncol & 63;
#pragma unroll
            for (int v = 0; v < 8; ++v) {
                const int m = m0 + mi * 16 + v + hi * 8;
                const float val = acc[mi][nt][v] + bb;
                if (mode == 2) {
                    Yf[(size_t)m * DD + ncol] = val;
                } else {
                    const int b = m >> 11, s = m & 2047;
                    size_t idx = (mode == 1)
                        ? ((size_t)((b * HH + hh) * HD + d) * SS + s)
                        : ((size_t)((b * HH + hh) * SS + s) * HD + d);
                    Yh[idx] = (_Float16)val;
                }
            }
        }
    }
}

// ---------------------------------------------------------------------------
// Flash attention: one wave per (b, h, 16-query block). Key blocks of 32.
// Q,K: [B,H,S,HD] f16, Vt: [B,H,HD,S] f16.  Output: [B,S,H,HD] f16.
// ---------------------------------------------------------------------------
__global__ __launch_bounds__(128) void attn_kernel(
    const _Float16* __restrict__ Q, const _Float16* __restrict__ K,
    const _Float16* __restrict__ Vt, _Float16* __restrict__ Vals)
{
    __shared__ _Float16 psh[4][16][40];   // per-wave P tile, 8-half padding

    const int lane = threadIdx.x & 31;
    const int wv   = threadIdx.x >> 5;
    const int task = blockIdx.x * 4 + wv;
    const int bh = task >> 7;            // 32 (b,h) pairs
    const int qb = task & 127;           // 128 query blocks of 16
    const int hi = lane >> 4, lm = lane & 15;

    const _Float16* Qb = Q  + (size_t)bh * SS * HD;
    const _Float16* Kb = K  + (size_t)bh * SS * HD;
    const _Float16* Vb = Vt + (size_t)bh * HD * SS;

    // Load Q fragments once: A-layout, contraction HD=64 -> two K=32 frags
    h16 qf[2];
#pragma unroll
    for (int f = 0; f < 2; ++f) {
        const _Float16* qr = Qb + (size_t)(qb * 16 + lm) * HD + f * 32;
        h8 a0 = *reinterpret_cast<const h8*>(qr + hi * 16);
        h8 a1 = *reinterpret_cast<const h8*>(qr + hi * 16 + 8);
        qf[f] = CAT16(a0, a1);
    }

    f8 o[4];
    float mrun[8], lrun[8];
#pragma unroll
    for (int n = 0; n < 4; ++n)
#pragma unroll
        for (int v = 0; v < 8; ++v) o[n][v] = 0.0f;
#pragma unroll
    for (int v = 0; v < 8; ++v) { mrun[v] = -1e30f; lrun[v] = 0.0f; }

    for (int j = 0; j < SS / 32; ++j) {
        // ---- batch K-fragment loads for both 16-key tiles ----
        const _Float16* kr0 = Kb + (size_t)(j * 32 + lm) * HD;
        const _Float16* kr1 = Kb + (size_t)(j * 32 + 16 + lm) * HD;
        h16 k00 = *reinterpret_cast<const h16*>(kr0 + hi * 16);
        h16 k01 = *reinterpret_cast<const h16*>(kr0 + 32 + hi * 16);
        h16 k10 = *reinterpret_cast<const h16*>(kr1 + hi * 16);
        h16 k11 = *reinterpret_cast<const h16*>(kr1 + 32 + hi * 16);
        __builtin_prefetch(kr0 + 32 * HD, 0, 3);
        __builtin_prefetch(kr1 + 32 * HD, 0, 3);

        f8 s[2];
#pragma unroll
        for (int v = 0; v < 8; ++v) { s[0][v] = 0.0f; s[1][v] = 0.0f; }
        s[0] = WMMA_F32_F16(qf[0], k00, s[0]);
        s[0] = WMMA_F32_F16(qf[1], k01, s[0]);
        s[1] = WMMA_F32_F16(qf[0], k10, s[1]);
        s[1] = WMMA_F32_F16(qf[1], k11, s[1]);
#pragma unroll
        for (int t = 0; t < 2; ++t)
#pragma unroll
            for (int v = 0; v < 8; ++v) s[t][v] *= 0.125f;   // 1/sqrt(64)

        // ---- online softmax: row reductions within 16-lane halves ----
        float mnew[8], alpha[8];
        f8 p0, p1;
#pragma unroll
        for (int v = 0; v < 8; ++v) {
            float r = fmaxf(s[0][v], s[1][v]);
            r = fmaxf(r, __shfl_xor(r, 1));
            r = fmaxf(r, __shfl_xor(r, 2));
            r = fmaxf(r, __shfl_xor(r, 4));
            r = fmaxf(r, __shfl_xor(r, 8));
            mnew[v]  = fmaxf(mrun[v], r);
            alpha[v] = __expf(mrun[v] - mnew[v]);
            p0[v] = __expf(s[0][v] - mnew[v]);
            p1[v] = __expf(s[1][v] - mnew[v]);
            float rs = p0[v] + p1[v];
            rs += __shfl_xor(rs, 1);
            rs += __shfl_xor(rs, 2);
            rs += __shfl_xor(rs, 4);
            rs += __shfl_xor(rs, 8);
            lrun[v] = lrun[v] * alpha[v] + rs;
            mrun[v] = mnew[v];
        }
#pragma unroll
        for (int n = 0; n < 4; ++n)
#pragma unroll
            for (int v = 0; v < 8; ++v) o[n][v] *= alpha[v];

        // ---- P (D-layout f32) -> LDS -> A-layout f16 fragment ----
#pragma unroll
        for (int v = 0; v < 8; ++v) {
            psh[wv][v + hi * 8][lm]      = (_Float16)p0[v];
            psh[wv][v + hi * 8][16 + lm] = (_Float16)p1[v];
        }
        asm volatile("s_wait_dscnt 0" ::: "memory");
        h8 pa0 = *reinterpret_cast<const h8*>(&psh[wv][lm][hi * 16]);
        h8 pa1 = *reinterpret_cast<const h8*>(&psh[wv][lm][hi * 16 + 8]);
        h16 pf = CAT16(pa0, pa1);

        // ---- O += P x V : contraction over 32 keys, 4 HD tiles ----
        h16 vf[4];
#pragma unroll
        for (int n = 0; n < 4; ++n)
            vf[n] = *reinterpret_cast<const h16*>(
                Vb + (size_t)(n * 16 + lm) * SS + j * 32 + hi * 16);
#pragma unroll
        for (int n = 0; n < 4; ++n)
            o[n] = WMMA_F32_F16(pf, vf[n], o[n]);
    }

    // ---- finalize and store to values [B,S,H,HD] ----
    const int b = bh >> 4, h = bh & 15;
#pragma unroll
    for (int n = 0; n < 4; ++n)
#pragma unroll
        for (int v = 0; v < 8; ++v) {
            const int srow = qb * 16 + v + hi * 8;
            const float val = o[n][v] / lrun[v];
            Vals[((size_t)(b * SS + srow) * HH + h) * HD + n * 16 + lm] = (_Float16)val;
        }
}

// ---------------------------------------------------------------------------
extern "C" void kernel_launch(void* const* d_in, const int* in_sizes, int n_in,
                              void* d_out, int out_size, void* d_ws, size_t ws_size,
                              hipStream_t stream) {
    // setup_inputs order: xq, xv, xk, Wq, bq, Wk, bk, Wv, bv, Wo, bo
    const float* xq = (const float*)d_in[0];
    const float* xv = (const float*)d_in[1];
    const float* xk = (const float*)d_in[2];
    const float* Wq = (const float*)d_in[3];
    const float* bq = (const float*)d_in[4];
    const float* Wk = (const float*)d_in[5];
    const float* bk = (const float*)d_in[6];
    const float* Wv = (const float*)d_in[7];
    const float* bv = (const float*)d_in[8];
    const float* Wo = (const float*)d_in[9];
    const float* bo = (const float*)d_in[10];

    const size_t ne_x = (size_t)MT * DD;        // 4.19M halves
    const size_t ne_w = (size_t)DD * DD;        // 1.05M halves
    _Float16* x16  = (_Float16*)d_ws;           // reused for xq/xk/xv
    _Float16* wq16 = x16  + ne_x;
    _Float16* wk16 = wq16 + ne_w;
    _Float16* wv16 = wk16 + ne_w;
    _Float16* wo16 = wv16 + ne_w;
    _Float16* q16  = wo16 + ne_w;
    _Float16* k16  = q16  + ne_x;
    _Float16* vt16 = k16  + ne_x;
    _Float16* vals = vt16 + ne_x;               // total 48 MB

    const int n4x = (int)(ne_x / 4), n4w = (int)(ne_w / 4);

    // Weight conversions (f32 -> f16)
    cvt_f32_f16<<<512, 256, 0, stream>>>(Wq, wq16, n4w);
    cvt_f32_f16<<<512, 256, 0, stream>>>(Wk, wk16, n4w);
    cvt_f32_f16<<<512, 256, 0, stream>>>(Wv, wv16, n4w);
    cvt_f32_f16<<<512, 256, 0, stream>>>(Wo, wo16, n4w);

    // Q projection
    cvt_f32_f16<<<1024, 256, 0, stream>>>(xq, x16, n4x);
    gemm16_kernel<<<512, 128, 0, stream>>>(x16, wq16, bq, q16, nullptr, 0);
    // K projection
    cvt_f32_f16<<<1024, 256, 0, stream>>>(xk, x16, n4x);
    gemm16_kernel<<<512, 128, 0, stream>>>(x16, wk16, bk, k16, nullptr, 0);
    // V projection (transposed output)
    cvt_f32_f16<<<1024, 256, 0, stream>>>(xv, x16, n4x);
    gemm16_kernel<<<512, 128, 0, stream>>>(x16, wv16, bv, vt16, nullptr, 1);

    // Attention: 32 (b,h) * 128 query-blocks = 4096 wave tasks
    attn_kernel<<<1024, 128, 0, stream>>>(q16, k16, vt16, vals);

    // Output projection -> f32 d_out
    gemm16_kernel<<<512, 128, 0, stream>>>(vals, wo16, bo, nullptr, (float*)d_out, 2);
}
